// DragonHGT_19602230739991
// MI455X (gfx1250) — matrix operand order repaired and checked by reading, not compiled, not run on hardware
//
#include <hip/hip_runtime.h>
#include <hip/hip_bf16.h>
#include <math.h>

#define H 8
#define D 16
#define C 128
#define MROWS 128   // rows per GEMM block

typedef __bf16 v16bf __attribute__((ext_vector_type(16)));
typedef float  v8f   __attribute__((ext_vector_type(8)));

__device__ __forceinline__ float gelu_exact(float x) {
    return 0.5f * x * (1.0f + erff(x * 0.70710678118654752f));
}

// ------------------- fill kernels -------------------
__global__ void fill_f32(float* __restrict__ p, float v, int n) {
    int i = blockIdx.x * blockDim.x + threadIdx.x;
    if (i < n) p[i] = v;
}
__global__ void fill_u32(unsigned* __restrict__ p, unsigned v, int n) {
    int i = blockIdx.x * blockDim.x + threadIdx.x;
    if (i < n) p[i] = v;
}

// ------------------- fuse per-relation head transform into weight -------------------
// Wdst[c, h*16+e] = sum_d Wsrc[c, h*16+d] * rel[h,d,e];  row c==C computes fused bias.
__global__ void fuse_rel(const float* __restrict__ Wsrc, const float* __restrict__ bsrc,
                         const float* __restrict__ rel,
                         float* __restrict__ Wdst, float* __restrict__ bdst) {
    int tid = blockIdx.x * blockDim.x + threadIdx.x;
    if (tid >= (C + 1) * C) return;
    int c = tid >> 7;        // 0..C (row C == bias)
    int o = tid & 127;       // h*16+e
    int h = o >> 4;
    const float* rh = rel + h * D * D + (o & 15);   // rel[h, d, e], stride D over d
    float acc = 0.f;
    if (c < C) {
        const float* wr = Wsrc + c * C + h * 16;
        #pragma unroll
        for (int d = 0; d < D; ++d) acc += wr[d] * rh[d * D];
        Wdst[c * C + o] = acc;
    } else {
        const float* br = bsrc + h * 16;
        #pragma unroll
        for (int d = 0; d < D; ++d) acc += br[d] * rh[d * D];
        bdst[o] = acc;
    }
}

// ------------------- pack fp32 weight [K=128,N=128] into WMMA bf16 B-fragments -------------------
// packed[((kt*8 + nt)*32 + lane)*16 + i] = B[kt*32 + (lane<16?0:16) + i, nt*16 + (lane&15)]
__global__ void pack_b(const float* __restrict__ W, __bf16* __restrict__ P) {
    int tid = blockIdx.x * blockDim.x + threadIdx.x;
    if (tid >= C * C) return;
    int i    = tid & 15;
    int lane = (tid >> 4) & 31;
    int nt   = (tid >> 9) & 7;
    int kt   = tid >> 12;
    int k = kt * 32 + ((lane < 16) ? 0 : 16) + i;
    int n = nt * 16 + (lane & 15);
    P[tid] = (__bf16)W[k * C + n];
}

// ------------------- WMMA GEMM: Y[M,128] = op(X)[M,128] @ B + bias (+epilogue) -------------------
// block = 256 threads = 8 waves. Block covers 128 rows; wave w owns 16-wide column tile w.
// A tile (128x128 fp32, contiguous in memory) is staged into LDS with async copies once per
// block; each wave keeps its 4 packed B fragments in registers and does up to 32 WMMAs.
// M is a multiple of 16, so tiles are all-valid or all-invalid -> uniform loop bound, no
// per-lane store predication (keeps EXEC all-ones throughout, as WMMA requires).
// MODE 0: Y = X@B + bias
// MODE 1: Y = beta*(gelu(X)@B + bias) + (1-beta)*skipsrc,  beta = sigmoid(*skipgate)
template <int MODE>
__global__ void gemm_wmma(const float* __restrict__ X, const __bf16* __restrict__ P,
                          const float* __restrict__ bias, float* __restrict__ Y,
                          const float* __restrict__ skipsrc, const float* __restrict__ skipgate,
                          int M) {
    __shared__ float lds[MROWS * C];           // 64 KB fp32 tile
    const int lane = threadIdx.x & 31;
    const int wave = threadIdx.x >> 5;
    const int m0   = blockIdx.x * MROWS;

    // ---- async-stage the contiguous 128x128 fp32 tile into LDS (16 B per lane per op) ----
    {
        const size_t max_off = (size_t)M * C - 4;          // last in-bounds float4 start
        #pragma unroll
        for (int j = 0; j < (MROWS * C) / (256 * 4); ++j) { // 16 iterations
            size_t fo = (size_t)(j * 256 + threadIdx.x) * 4;   // float index in tile
            size_t go = (size_t)m0 * C + fo;                   // float index in X
            if (go > max_off) go = max_off;                    // clamp for tail block
            unsigned ldst = (unsigned)(fo * 4);                // LDS byte offset (array at 0)
            unsigned long long gp = (unsigned long long)(uintptr_t)(X + go);
            asm volatile("global_load_async_to_lds_b128 %0, %1, off"
                         :: "v"(ldst), "v"(gp) : "memory");
        }
        asm volatile("s_wait_asynccnt 0" ::: "memory");
    }
    __syncthreads();

    // ---- load this wave's 4 B fragments once (stay in registers) ----
    v16bf bfrag[4];
    #pragma unroll
    for (int kt = 0; kt < 4; ++kt)
        bfrag[kt] = *(const v16bf*)(P + (((kt * 8 + wave) * 32 + lane) << 4));

    const int lr  = lane & 15;
    const int kb  = (lane < 16) ? 0 : 16;    // A-layout K base per lane half
    const int col = wave * 16 + lr;
    const float bv = bias[col];
    float beta = 0.f;
    if (MODE == 1) beta = 1.f / (1.f + __expf(-skipgate[0]));

    const int mtEnd = min(MROWS / 16, (M - m0) >> 4);   // uniform across block
    #pragma unroll 2
    for (int mt = 0; mt < mtEnd; ++mt) {
        v8f acc = {};
        #pragma unroll
        for (int kt = 0; kt < 4; ++kt) {
            const float* ap = &lds[(mt * 16 + lr) * C + kt * 32 + kb];
            v16bf a;
            #pragma unroll
            for (int i = 0; i < 8; ++i) {
                float f0 = ap[i];         // K = kt*32 + kb + i
                float f1 = ap[8 + i];     // K = kt*32 + kb + 8 + i
                if (MODE == 1) { f0 = gelu_exact(f0); f1 = gelu_exact(f1); }
                a[i]     = (__bf16)f0;
                a[8 + i] = (__bf16)f1;
            }
            acc = __builtin_amdgcn_wmma_f32_16x16x32_bf16(false, a, false, bfrag[kt],
                                                          (short)0, acc, false, false);
        }
        const int r0 = m0 + mt * 16 + ((lane < 16) ? 0 : 8);
        float* yp = Y + (size_t)r0 * C + col;           // 8 stores at immediate offsets r*512B
        if (MODE == 0) {
            #pragma unroll
            for (int r = 0; r < 8; ++r) yp[r * C] = acc[r] + bv;
        } else {
            const float* sp = skipsrc + (size_t)r0 * C + col;
            #pragma unroll
            for (int r = 0; r < 8; ++r) {
                float o = acc[r] + bv;
                yp[r * C] = beta * o + (1.f - beta) * sp[r * C];
            }
        }
    }
}

// ------------------- edge kernels -------------------
__device__ __forceinline__ unsigned f32_key(float f) {
    unsigned b = __float_as_uint(f);
    return (b & 0x80000000u) ? ~b : (b | 0x80000000u);
}
__device__ __forceinline__ float key_f32(unsigned k) {
    unsigned b = (k & 0x80000000u) ? (k & 0x7fffffffu) : ~k;
    return __uint_as_float(b);
}

__global__ void edge_logit(const int* __restrict__ src, const int* __restrict__ dst,
                           const float* __restrict__ q, const float* __restrict__ kr,
                           const float* __restrict__ prel, float* __restrict__ lg,
                           unsigned* __restrict__ mmax, int E) {
    int t = blockIdx.x * blockDim.x + threadIdx.x;
    if (t >= E * H) return;
    int e = t >> 3, h = t & 7;
    int s = src[e], d = dst[e];
    const float4* qp = (const float4*)(q + (size_t)d * C + h * D);
    const float4* kp = (const float4*)(kr + (size_t)s * C + h * D);
    float acc = 0.f;
    #pragma unroll
    for (int i = 0; i < 4; ++i) {
        float4 a = qp[i], b = kp[i];
        acc += a.x * b.x + a.y * b.y + a.z * b.z + a.w * b.w;
    }
    float v = acc * prel[h] * 0.25f;   // scale = 1/sqrt(D) = 0.25
    lg[t] = v;
    atomicMax(mmax + (size_t)d * H + h, f32_key(v));
}

__global__ void edge_exp(const int* __restrict__ dst, float* __restrict__ lg,
                         const unsigned* __restrict__ mmax, float* __restrict__ ssum, int E) {
    int t = blockIdx.x * blockDim.x + threadIdx.x;
    if (t >= E * H) return;
    int e = t >> 3, h = t & 7;
    int d = dst[e];
    float m = key_f32(mmax[(size_t)d * H + h]);
    if (!isfinite(m)) m = 0.f;
    float w = __expf(lg[t] - m);
    lg[t] = w;
    atomicAdd(ssum + (size_t)d * H + h, w);
}

__global__ void edge_scatter(const int* __restrict__ src, const int* __restrict__ dst,
                             const float* __restrict__ lg, const float* __restrict__ ssum,
                             const float* __restrict__ vr, float* __restrict__ agg, int E) {
    int t = blockIdx.x * blockDim.x + threadIdx.x;
    if (t >= E * H) return;
    int e = t >> 3, h = t & 7;
    int s = src[e], d = dst[e];
    float alpha = lg[t] / (ssum[(size_t)d * H + h] + 1e-16f);
    const float* vp = vr + (size_t)s * C + h * D;
    float* ap = agg + (size_t)d * C + h * D;
    #pragma unroll
    for (int i = 0; i < D; ++i) atomicAdd(ap + i, vp[i] * alpha);
}

// ------------------- launch -------------------
extern "C" void kernel_launch(void* const* d_in, const int* in_sizes, int n_in,
                              void* d_out, int out_size, void* d_ws, size_t ws_size,
                              hipStream_t stream) {
    const float* xA   = (const float*)d_in[0];
    const float* xB   = (const float*)d_in[1];
    const int*   eAB  = (const int*)d_in[2];
    const int*   eBA  = (const int*)d_in[3];
    const int*   eAA  = (const int*)d_in[4];
    const float* Wk   = (const float*)d_in[5];
    const float* bk   = (const float*)d_in[6];
    const float* Wq   = (const float*)d_in[7];
    const float* bq   = (const float*)d_in[8];
    const float* Wv   = (const float*)d_in[9];
    const float* bv   = (const float*)d_in[10];
    const float* Wa   = (const float*)d_in[11];
    const float* ba   = (const float*)d_in[12];
    const float* skip = (const float*)d_in[13];
    const float* a_rel = (const float*)d_in[14];
    const float* m_rel = (const float*)d_in[15];
    const float* p_rel = (const float*)d_in[16];

    const int N = in_sizes[0] / C;   // 100000 (multiple of 16)
    const int E = in_sizes[2] / 2;   // 800000
    float* out = (float*)d_out;
    const float* xs[2] = {xA, xB};

    // carve workspace (~360 MB)
    char* w = (char*)d_ws;
    auto carve = [&](size_t bytes) { char* p = w; w += (bytes + 255) & ~(size_t)255; return p; };
    float*    qbuf  = (float*)carve(2ull * N * C * sizeof(float));
    float*    krbuf = (float*)carve((size_t)N * C * sizeof(float));
    float*    vrbuf = (float*)carve((size_t)N * C * sizeof(float));
    float*    agg   = (float*)carve(2ull * N * C * sizeof(float));
    float*    lg    = (float*)carve((size_t)E * H * sizeof(float));
    unsigned* mmax  = (unsigned*)carve((size_t)N * H * sizeof(unsigned));
    float*    ssum  = (float*)carve((size_t)N * H * sizeof(float));
    float*    fW    = (float*)carve((size_t)C * C * sizeof(float));
    float*    fB    = (float*)carve(C * sizeof(float));
    __bf16*   pq0   = (__bf16*)carve((size_t)C * C * sizeof(__bf16));
    __bf16*   pq1   = (__bf16*)carve((size_t)C * C * sizeof(__bf16));
    __bf16*   pk    = (__bf16*)carve((size_t)C * C * sizeof(__bf16));
    __bf16*   pv    = (__bf16*)carve((size_t)C * C * sizeof(__bf16));
    __bf16*   pa0   = (__bf16*)carve((size_t)C * C * sizeof(__bf16));
    __bf16*   pa1   = (__bf16*)carve((size_t)C * C * sizeof(__bf16));
    __bf16*   pq[2] = {pq0, pq1};
    __bf16*   pa[2] = {pa0, pa1};
    (void)n_in; (void)out_size; (void)ws_size;

    const int TPB = 256;
    auto nb = [&](int n) { return (n + TPB - 1) / TPB; };
    const int gemmBlocks = (N + MROWS - 1) / MROWS;

    // zero aggregation buffer
    const int nAgg = 2 * N * C;
    fill_f32<<<nb(nAgg), TPB, 0, stream>>>(agg, 0.f, nAgg);

    // q projections (both node types)
    for (int t = 0; t < 2; ++t) {
        pack_b<<<nb(C * C), TPB, 0, stream>>>(Wq + t * C * C, pq[t]);
        gemm_wmma<0><<<gemmBlocks, 256, 0, stream>>>(xs[t], pq[t], bq + t * C,
                                                     qbuf + (size_t)t * N * C,
                                                     nullptr, nullptr, N);
    }

    struct Rel { int s, t, r; const int* e; };
    const Rel rels[3] = { {0, 1, 0, eAB}, {1, 0, 1, eBA}, {0, 0, 2, eAA} };
    const int EH = E * H;

    for (int i = 0; i < 3; ++i) {
        const Rel R = rels[i];
        const int* src = R.e;
        const int* dst = R.e + E;
        // kr = x[s] @ (Wk[s] folded with a_rel[r]) + fused bias
        fuse_rel<<<nb((C + 1) * C), TPB, 0, stream>>>(Wk + R.s * C * C, bk + R.s * C,
                                                      a_rel + R.r * H * D * D, fW, fB);
        pack_b<<<nb(C * C), TPB, 0, stream>>>(fW, pk);
        gemm_wmma<0><<<gemmBlocks, 256, 0, stream>>>(xs[R.s], pk, fB, krbuf,
                                                     nullptr, nullptr, N);
        // vr = x[s] @ (Wv[s] folded with m_rel[r]) + fused bias
        fuse_rel<<<nb((C + 1) * C), TPB, 0, stream>>>(Wv + R.s * C * C, bv + R.s * C,
                                                      m_rel + R.r * H * D * D, fW, fB);
        pack_b<<<nb(C * C), TPB, 0, stream>>>(fW, pv);
        gemm_wmma<0><<<gemmBlocks, 256, 0, stream>>>(xs[R.s], pv, fB, vrbuf,
                                                     nullptr, nullptr, N);
        // segment softmax over destination nodes + weighted scatter
        fill_u32<<<nb(N * H), TPB, 0, stream>>>(mmax, 0u, N * H);
        fill_f32<<<nb(N * H), TPB, 0, stream>>>(ssum, 0.f, N * H);
        edge_logit<<<nb(EH), TPB, 0, stream>>>(src, dst, qbuf + (size_t)R.t * N * C, krbuf,
                                               p_rel + R.r * H, lg, mmax, E);
        edge_exp<<<nb(EH), TPB, 0, stream>>>(dst, lg, mmax, ssum, E);
        edge_scatter<<<nb(EH), TPB, 0, stream>>>(src, dst, lg, ssum, vrbuf,
                                                 agg + (size_t)R.t * N * C, E);
    }

    // output projection: out = sigmoid(skip)*[gelu(agg)@Wa + ba] + (1-sigmoid(skip))*x
    for (int t = 0; t < 2; ++t) {
        pack_b<<<nb(C * C), TPB, 0, stream>>>(Wa + t * C * C, pa[t]);
        gemm_wmma<1><<<gemmBlocks, 256, 0, stream>>>(agg + (size_t)t * N * C, pa[t], ba + t * C,
                                                     out + (size_t)t * N * C, xs[t], skip + t, N);
    }
}
